// CapsuleNetwork_46626164966083
// MI455X (gfx1250) — matrix operand order repaired and checked by reading, not compiled, not run on hardware
//
#include <hip/hip_runtime.h>

// CapsuleNetwork forward for MI455X (gfx1250, wave32).
// Both convs run on v_wmma_f32_16x16x32_bf16:
//   conv1:  GEMM M=256 x N=51200 x K=96 (81 taps zero-padded), im2col staged.
//   pcaps:  implicit GEMM M=256 x N=4608 x K=20736 (NHWC bf16 activations).
// u_hat batched GEMM also uses WMMA (K=8 padded to 32). u_hat kept bf16
// (47 MB, resident in the 192 MB L2); routing is L2-bandwidth bound VALU.

typedef __attribute__((ext_vector_type(16))) __bf16        v16bf;
typedef __attribute__((ext_vector_type(8)))  float         v8f;
typedef __attribute__((ext_vector_type(4)))  unsigned int  u32x4;

union Frag32 {
    u32x4          q[2];
    unsigned short h[16];
    v16bf          v;
};

__device__ __forceinline__ unsigned short f2bf(float f) {
    unsigned int u = __float_as_uint(f);
    u += 0x7FFFu + ((u >> 16) & 1u);          // round-to-nearest-even
    return (unsigned short)(u >> 16);
}
__device__ __forceinline__ float bf2f(unsigned short h) {
    return __uint_as_float(((unsigned int)h) << 16);
}

// A-fragment k order chosen by this code for 16x32 bf16 (contiguous; B uses the same):
// lane<16 holds k in {0..15}, lane>=16 holds k in {16..31}.
__device__ __forceinline__ int afrag_k(int lane, int e) {
    int klo = (lane & 16) ? 16 : 0;
    return klo + e;
}

// ---------------------------------------------------------------------------
// im2col for conv1: imgcol[n][k] bf16, n = b*400 + y*20 + x (51200), k = 96
// (k = ky*9+kx for k<81, zero-padded above). One element per thread.
__global__ void k_im2col(const float* __restrict__ img,
                         unsigned short* __restrict__ imgcol) {
    int idx = blockIdx.x * 256 + threadIdx.x;   // exact: 19200*256 = 4915200
    int k = idx % 96;
    int n = idx / 96;
    unsigned short v = 0;
    if (k < 81) {
        int b = n / 400;
        int pos = n % 400;
        int y = pos / 20, x = pos % 20;
        int ky = k / 9, kx = k % 9;
        v = f2bf(img[(size_t)b * 784 + (y + ky) * 28 + (x + kx)]);
    }
    imgcol[idx] = v;
}

// Pack conv1 weights into A-fragment order: c1p[(kblk*16+mt)*512 + lane*16 + e]
__global__ void k_pack_c1w(const float* __restrict__ w,
                           unsigned short* __restrict__ c1p) {
    int idx  = blockIdx.x * 256 + threadIdx.x;   // exact: 96*256 = 24576
    int e    = idx & 15;
    int lane = (idx >> 4) & 31;
    int mt   = (idx >> 9) & 15;
    int kblk = idx >> 13;                        // 0..2
    int m    = mt * 16 + (lane & 15);
    int k    = kblk * 32 + afrag_k(lane, e);
    c1p[idx] = (k < 81) ? f2bf(w[(size_t)m * 81 + k]) : (unsigned short)0;
}

// conv1 GEMM + bias + ReLU -> x NHWC bf16 [51200][256]
__global__ void __launch_bounds__(128)
k_conv1_wmma(const unsigned short* __restrict__ imgcol,
             const unsigned short* __restrict__ c1p,
             const float* __restrict__ bias,
             unsigned short* __restrict__ xb) {
    int wave = blockIdx.x * 4 + (threadIdx.x >> 5);  // < 51200
    int lane = threadIdx.x & 31;
    int mt = wave & 15;           // 16 M tiles
    int nt = wave >> 4;           // 3200 N tiles
    int half16 = (lane >> 4) & 1;
    int nl = nt * 16 + (lane & 15);

    const unsigned short* apb = c1p + (size_t)(mt * 32 + lane) * 16;
    const unsigned short* bpb = imgcol + (size_t)nl * 96 + half16 * 16;

    v8f acc = {0.f, 0.f, 0.f, 0.f, 0.f, 0.f, 0.f, 0.f};
#pragma unroll
    for (int kblk = 0; kblk < 3; ++kblk) {
        Frag32 fa, fb;
        const u32x4* a4 = (const u32x4*)(apb + (size_t)kblk * 8192);
        fa.q[0] = a4[0];
        fa.q[1] = a4[1];
        const u32x4* b4 = (const u32x4*)(bpb + kblk * 32);
        fb.q[0] = b4[0];
        fb.q[1] = b4[1];
        acc = __builtin_amdgcn_wmma_f32_16x16x32_bf16(
            false, fa.v, false, fb.v, (short)0, acc, false, false);
    }
#pragma unroll
    for (int r = 0; r < 8; ++r) {
        int oc = mt * 16 + r + half16 * 8;
        float val = fmaxf(acc[r] + bias[oc], 0.0f);
        xb[(size_t)nl * 256 + oc] = f2bf(val);
    }
}

// ---------------------------------------------------------------------------
// Pack pcaps weights into WMMA A-fragment order:
// ap[((p*8+kblk)*16+mt)*512 + lane*16 + e]  (bf16), 5,308,416 elements.
__global__ void k_pack_w(const float* __restrict__ pw,
                         unsigned short* __restrict__ ap) {
    int idx  = blockIdx.x * 256 + threadIdx.x;   // exact: 20736*256
    int e    = idx & 15;
    int lane = (idx >> 4) & 31;
    int mt   = (idx >> 9) & 15;
    int kblk = (idx >> 13) & 7;
    int p    = idx >> 16;                        // 0..80
    int m    = mt * 16 + (lane & 15);
    int ic   = kblk * 32 + afrag_k(lane, e);
    int ky = p / 9, kx = p % 9;
    ap[idx] = f2bf(pw[(size_t)(m * 256 + ic) * 81 + ky * 9 + kx]);
}

// ---------------------------------------------------------------------------
// Primary-caps conv as implicit GEMM: C[256 oc][4608 pos], K = 81 taps x 256 ic.
// One wave per 16x16 tile; 648 WMMAs per wave. Output written as
// u_pre[b][i=cap*36+s][d] fp32.
__global__ void __launch_bounds__(128)
k_pcaps(const unsigned short* __restrict__ xb,   // NHWC bf16
        const unsigned short* __restrict__ ap,   // packed A frags
        const float* __restrict__ pbias,         // [256]
        float* __restrict__ u_pre) {             // [128][1152][8]
    int wave = blockIdx.x * 4 + (threadIdx.x >> 5);
    int lane = threadIdx.x & 31;
    int mt = wave & 15;          // 16 M tiles (out channels)
    int nt = wave >> 4;          // 288 N tiles (positions)
    int half16 = (lane >> 4) & 1;

    // B-fragment lane mapping: n = lane&15, k half selected by lane>=16
    int nl = nt * 16 + (lane & 15);
    int b  = nl / 36, s = nl % 36;
    int oy = s / 6,  ox = s % 6;
    const unsigned short* xrow0 = xb + (size_t)b * 400 * 256;
    int koff = half16 * 16;

    const unsigned short* apb = ap + (size_t)(mt * 32 + lane) * 16;

    v8f acc = {0.f, 0.f, 0.f, 0.f, 0.f, 0.f, 0.f, 0.f};
    for (int p = 0; p < 81; ++p) {
        int ky = p / 9, kx = p % 9;
        int rowoff = ((2 * oy + ky) * 20 + (2 * ox + kx)) * 256;
        const unsigned short* bptr0 = xrow0 + rowoff + koff;
        const unsigned short* aptr0 = apb + (size_t)p * 65536;
#pragma unroll
        for (int kblk = 0; kblk < 8; ++kblk) {
            Frag32 fa, fb;
            const u32x4* a4 = (const u32x4*)(aptr0 + kblk * 8192);
            fa.q[0] = a4[0];
            fa.q[1] = a4[1];
            const u32x4* b4 = (const u32x4*)(bptr0 + kblk * 32);
            fb.q[0] = b4[0];
            fb.q[1] = b4[1];
            acc = __builtin_amdgcn_wmma_f32_16x16x32_bf16(
                false, fa.v, false, fb.v, (short)0, acc, false, false);
        }
    }
    // C/D layout: vgpr r, lane l -> M = r + (l>=16)*8, N = l&15
#pragma unroll
    for (int r = 0; r < 8; ++r) {
        int m = mt * 16 + r + half16 * 8;        // out channel
        float val = acc[r] + pbias[m];
        int i = (m >> 3) * 36 + s;               // capsule-major index
        u_pre[((size_t)b * 1152 + i) * 8 + (m & 7)] = val;
    }
}

// ---------------------------------------------------------------------------
// squash over d=8 -> u bf16 [128][1152][8]
__global__ void k_squash_u(const float* __restrict__ u_pre,
                           unsigned short* __restrict__ ub) {
    int idx = blockIdx.x * 256 + threadIdx.x;    // exact: 576*256 = 147456
    const float* p = u_pre + (size_t)idx * 8;
    float v[8], sq = 0.f;
#pragma unroll
    for (int d = 0; d < 8; ++d) { v[d] = p[d]; sq += v[d] * v[d]; }
    float sc = sqrtf(sq) / (1.0f + sq);
#pragma unroll
    for (int d = 0; d < 8; ++d) ub[(size_t)idx * 8 + d] = f2bf(v[d] * sc);
}

// ---------------------------------------------------------------------------
// u_hat[b,o,i,d] = sum_c u[b,i,c] * route_w[o,i,c,d]  via WMMA (K=8 padded).
// One wave per (o, i, btile of 16 batches). Lanes 0-15 carry all valid K.
__global__ void __launch_bounds__(128)
k_uhat(const unsigned short* __restrict__ ub,    // [128][1152][8] bf16
       const float* __restrict__ rw,             // [10][1152][8][16] f32
       unsigned short* __restrict__ uhat) {      // [128][10][1152][16] bf16
    int wave = blockIdx.x * 4 + (threadIdx.x >> 5);  // < 92160
    int lane = threadIdx.x & 31;
    int btile = wave & 7;
    int rest  = wave >> 3;
    int i = rest % 1152;
    int o = rest / 1152;

    u32x4 z = {0u, 0u, 0u, 0u};
    Frag32 fa, fb;
    fa.q[0] = z; fa.q[1] = z;
    fb.q[0] = z; fb.q[1] = z;
    if (lane < 16) {
        int bb = btile * 16 + lane;              // A row m = batch
        fa.q[0] = *(const u32x4*)(ub + ((size_t)bb * 1152 + i) * 8);
        int d = lane;                            // B col n = d
        const float* wp = rw + ((size_t)(o * 1152 + i) * 8) * 16 + d;
#pragma unroll
        for (int c = 0; c < 8; ++c) fb.h[c] = f2bf(wp[c * 16]);
    }
    v8f acc = {0.f, 0.f, 0.f, 0.f, 0.f, 0.f, 0.f, 0.f};
    acc = __builtin_amdgcn_wmma_f32_16x16x32_bf16(
        false, fa.v, false, fb.v, (short)0, acc, false, false);

    int half16 = (lane >> 4) & 1;
    int d = lane & 15;
#pragma unroll
    for (int r = 0; r < 8; ++r) {
        int bb = btile * 16 + r + half16 * 8;
        uhat[(((size_t)bb * 10 + o) * 1152 + i) * 16 + d] = f2bf(acc[r]);
    }
}

// ---------------------------------------------------------------------------
__global__ void k_zero(float* __restrict__ p, int n) {
    int i = blockIdx.x * 256 + threadIdx.x;
    if (i < n) p[i] = 0.0f;
}

// softmax over i (1152) for each (b,o); logits/probs layout [b][o][i]
__global__ void k_softmax(const float* __restrict__ logits,
                          float* __restrict__ probs) {
    int bo = blockIdx.x;                 // 1280 = 128*10
    int t  = threadIdx.x;                // 256
    const float* lp = logits + (size_t)bo * 1152;
    __shared__ float red[256];
    float mx = -1e30f;
    for (int i = t; i < 1152; i += 256) mx = fmaxf(mx, lp[i]);
    red[t] = mx;
    for (int st = 128; st > 0; st >>= 1) {
        __syncthreads();
        if (t < st) red[t] = fmaxf(red[t], red[t + st]);
    }
    __syncthreads();
    mx = red[0];
    __syncthreads();
    float sum = 0.f;
    for (int i = t; i < 1152; i += 256) sum += expf(lp[i] - mx);
    red[t] = sum;
    for (int st = 128; st > 0; st >>= 1) {
        __syncthreads();
        if (t < st) red[t] += red[t + st];
    }
    __syncthreads();
    float inv = 1.0f / red[0];
    float* pp = probs + (size_t)bo * 1152;
    for (int i = t; i < 1152; i += 256) pp[i] = expf(lp[i] - mx) * inv;
}

// s[b,o,d] = sum_i uhat[b,o,i,d] * probs[b,o,i]; outputs = squash(s)
__global__ void k_s_squash(const float* __restrict__ probs,
                           const unsigned short* __restrict__ uhat,
                           float* __restrict__ outs,
                           float* __restrict__ dout) {
    int bo = blockIdx.x;
    int t  = threadIdx.x;
    const float* pp = probs + (size_t)bo * 1152;
    const unsigned short* up = uhat + (size_t)bo * 1152 * 16;
    float part[16];
#pragma unroll
    for (int d = 0; d < 16; ++d) part[d] = 0.f;
    for (int i = t; i < 1152; i += 256) {
        float pv = pp[i];
        const u32x4* h4 = (const u32x4*)(up + (size_t)i * 16);
        Frag32 f; f.q[0] = h4[0]; f.q[1] = h4[1];
#pragma unroll
        for (int d = 0; d < 16; ++d) part[d] = fmaf(pv, bf2f(f.h[d]), part[d]);
    }
    __shared__ float red[256 * 16];
#pragma unroll
    for (int d = 0; d < 16; ++d) red[t * 16 + d] = part[d];
    for (int st = 128; st > 0; st >>= 1) {
        __syncthreads();
        if (t < st)
            for (int d = 0; d < 16; ++d) red[t * 16 + d] += red[(t + st) * 16 + d];
    }
    __syncthreads();
    if (t == 0) {
        float sq = 0.f;
        for (int d = 0; d < 16; ++d) sq += red[d] * red[d];
        float sc = sqrtf(sq) / (1.0f + sq);
        for (int d = 0; d < 16; ++d) {
            float v = red[d] * sc;
            outs[(size_t)bo * 16 + d] = v;
            dout[(size_t)bo * 16 + d] = v;
        }
    }
}

// logits[b,o,i] += sum_d outs[b,o,d] * uhat[b,o,i,d]
__global__ void k_logits_update(const float* __restrict__ outs,
                                const unsigned short* __restrict__ uhat,
                                float* __restrict__ logits) {
    int tid = blockIdx.x * 256 + threadIdx.x;   // exact: 5760*256 = 1474560
    int i  = tid % 1152;
    int bo = tid / 1152;
    const float* op = outs + (size_t)bo * 16;
    const u32x4* h4 = (const u32x4*)(uhat + ((size_t)bo * 1152 + i) * 16);
    Frag32 f; f.q[0] = h4[0]; f.q[1] = h4[1];
    float dot = 0.f;
#pragma unroll
    for (int d = 0; d < 16; ++d) dot = fmaf(op[d], bf2f(f.h[d]), dot);
    logits[(size_t)bo * 1152 + i] += dot;
}

// ---------------------------------------------------------------------------
extern "C" void kernel_launch(void* const* d_in, const int* in_sizes, int n_in,
                              void* d_out, int out_size, void* d_ws, size_t ws_size,
                              hipStream_t stream) {
    const float* img = (const float*)d_in[0];   // [128,1,28,28]
    const float* c1w = (const float*)d_in[1];   // [256,1,9,9]
    const float* c1b = (const float*)d_in[2];   // [256]
    const float* pcw = (const float*)d_in[3];   // [32,8,256,9,9]
    const float* pcb = (const float*)d_in[4];   // [32,8]
    const float* rw  = (const float*)d_in[5];   // [10,1152,8,16]
    float* dout = (float*)d_out;                // [128,10,16]

    char* ws = (char*)d_ws;
    size_t off = 0;
    unsigned short* xb     = (unsigned short*)(ws + off); off += 26214400;  // x NHWC bf16
    unsigned short* ap     = (unsigned short*)(ws + off); off += 10616832;  // packed pcaps A
    float*          u_pre  = (float*)(ws + off);          off += 4718592;   // [128,1152,8]
    unsigned short* ub     = (unsigned short*)(ws + off); off += 2359296;   // u bf16
    unsigned short* uhat   = (unsigned short*)(ws + off); off += 47185920;  // u_hat bf16
    float*          logits = (float*)(ws + off);          off += 5898240;   // [b][o][i]
    float*          probs  = (float*)(ws + off);          off += 5898240;
    float*          outs   = (float*)(ws + off);          off += 81920;
    unsigned short* imgcol = (unsigned short*)(ws + off); off += 9830400;   // [51200][96]
    unsigned short* c1p    = (unsigned short*)(ws + off); off += 49152;     // packed conv1 A

    k_im2col<<<dim3(19200), dim3(256), 0, stream>>>(img, imgcol);
    k_pack_c1w<<<dim3(96), dim3(256), 0, stream>>>(c1w, c1p);
    k_conv1_wmma<<<dim3(12800), dim3(128), 0, stream>>>(imgcol, c1p, c1b, xb);
    k_pack_w<<<dim3(20736), dim3(256), 0, stream>>>(pcw, ap);
    k_pcaps<<<dim3(1152), dim3(128), 0, stream>>>(xb, ap, pcb, u_pre);
    k_squash_u<<<dim3(576), dim3(256), 0, stream>>>(u_pre, ub);
    k_uhat<<<dim3(23040), dim3(128), 0, stream>>>(ub, rw, uhat);
    k_zero<<<dim3(5760), dim3(256), 0, stream>>>(logits, 1474560);
    for (int it = 1; it <= 3; ++it) {
        k_softmax<<<dim3(1280), dim3(256), 0, stream>>>(logits, probs);
        k_s_squash<<<dim3(1280), dim3(256), 0, stream>>>(probs, uhat, outs, dout);
        if (it < 3)
            k_logits_update<<<dim3(5760), dim3(256), 0, stream>>>(outs, uhat, logits);
    }
}
